// Conv2d_uint8_custom_858993459359
// MI455X (gfx1250) — 8-bit matrix operand order repaired and checked by reading, compiled, not run on hardware
//
#include <hip/hip_runtime.h>


typedef int v8i __attribute__((ext_vector_type(8)));
typedef unsigned int v4u __attribute__((ext_vector_type(4)));
typedef unsigned long long u64;

#define C_IN    64
#define H_IN    56
#define W_IN    56
#define BATCH   16
#define O_OUT   128
#define KDIM    576
#define SPATIAL (H_IN * W_IN)
#define MROWS   (BATCH * SPATIAL)

__device__ __forceinline__ unsigned enc_f32(float f) {
  unsigned u = __float_as_uint(f);
  return (u & 0x80000000u) ? ~u : (u | 0x80000000u);
}
__device__ __forceinline__ float dec_f32(unsigned e) {
  return __uint_as_float((e & 0x80000000u) ? (e & 0x7FFFFFFFu) : ~e);
}
__device__ __forceinline__ unsigned umin32(unsigned a, unsigned b) { return a < b ? a : b; }
__device__ __forceinline__ unsigned umax32(unsigned a, unsigned b) { return a > b ? a : b; }

__global__ void init_slots(unsigned* slots) {
  if (threadIdx.x == 0) {
    slots[0] = 0xFFFFFFFFu; slots[1] = 0u;
    slots[2] = 0xFFFFFFFFu; slots[3] = 0u;
  }
}

__global__ void minmax_kernel(const float* __restrict__ x, int n,
                              unsigned* __restrict__ slots, int slotBase) {
  __shared__ unsigned smn[256], smx[256];
  unsigned lmn = 0xFFFFFFFFu, lmx = 0u;
  for (int i = blockIdx.x * blockDim.x + threadIdx.x; i < n; i += gridDim.x * blockDim.x) {
    unsigned e = enc_f32(x[i]);
    lmn = umin32(lmn, e); lmx = umax32(lmx, e);
  }
  smn[threadIdx.x] = lmn; smx[threadIdx.x] = lmx;
  __syncthreads();
  for (int s = 128; s > 0; s >>= 1) {
    if ((int)threadIdx.x < s) {
      smn[threadIdx.x] = umin32(smn[threadIdx.x], smn[threadIdx.x + s]);
      smx[threadIdx.x] = umax32(smx[threadIdx.x], smx[threadIdx.x + s]);
    }
    __syncthreads();
  }
  if (threadIdx.x == 0) {
    atomicMin(&slots[slotBase + 0], smn[0]);
    atomicMax(&slots[slotBase + 1], smx[0]);
  }
}

__global__ void compute_scales(const unsigned* __restrict__ slots, float* __restrict__ scales) {
  if (threadIdx.x == 0) {
    float mnx = dec_f32(slots[0]), mxx = dec_f32(slots[1]);
    float sf = (mxx - mnx) / 255.0f;
    scales[0] = sf;
    scales[1] = -rintf(mnx / sf);
    float mnw = dec_f32(slots[2]), mxw = dec_f32(slots[3]);
    float sw = (mxw - mnw) / 255.0f;
    scales[2] = sw;
    scales[3] = -rintf(mnw / sw);
  }
}

__global__ void quantize_weight(const float* __restrict__ w, const float* __restrict__ scales,
                                unsigned char* __restrict__ qw) {
  int i = blockIdx.x * blockDim.x + threadIdx.x;
  if (i < O_OUT * KDIM) {
    float q = fminf(fmaxf(rintf(w[i] / scales[2]) + scales[3], 0.0f), 255.0f);
    qw[i] = (unsigned char)q;
  }
}

__global__ void qw_row_sum(const unsigned char* __restrict__ qw, float* __restrict__ qwsum) {
  __shared__ unsigned acc[64];
  int o = blockIdx.x;
  unsigned s = 0;
  for (int j = threadIdx.x; j < KDIM; j += 64) s += (unsigned)qw[o * KDIM + j];
  acc[threadIdx.x] = s;
  __syncthreads();
  for (int st = 32; st > 0; st >>= 1) {
    if ((int)threadIdx.x < st) acc[threadIdx.x] += acc[threadIdx.x + st];
    __syncthreads();
  }
  if (threadIdx.x == 0) qwsum[o] = (float)acc[0];
}

__global__ void im2col_quant(const float* __restrict__ x, const float* __restrict__ scales,
                             unsigned char* __restrict__ patches) {
  int idx = blockIdx.x * blockDim.x + threadIdx.x;
  if (idx >= MROWS * KDIM) return;
  int m = idx / KDIM;
  int k = idx - m * KDIM;
  int c  = k / 9;  int rem = k - c * 9;
  int kh = rem / 3; int kw = rem - kh * 3;
  int b  = m / SPATIAL; int s = m - b * SPATIAL;
  int oh = s / W_IN;    int ow = s - oh * W_IN;
  int ih = oh + kh - 1, iw = ow + kw - 1;
  unsigned char q = 0;
  if ((unsigned)ih < (unsigned)H_IN && (unsigned)iw < (unsigned)W_IN) {
    float v = x[((b * C_IN + c) * H_IN + ih) * W_IN + iw];
    q = (unsigned char)fminf(fmaxf(rintf(v / scales[0]) + scales[1], 0.0f), 255.0f);
  }
  patches[idx] = q;
}


struct Frag { v8i a0, a1, b0, b1; };

__device__ __forceinline__ void load_slab(Frag& f,
    const unsigned char* __restrict__ aRow0, const unsigned char* __restrict__ aRow1,
    const unsigned char* __restrict__ bRow0, const unsigned char* __restrict__ bRow1,
    int k0) {
  union { v8i v; u64 d[4]; } A0, A1;
  const u64* p0 = (const u64*)(aRow0 + k0);
  const u64* p1 = (const u64*)(aRow1 + k0);
  A0.d[0] = p0[0]; A0.d[1] = p0[1]; A0.d[2] = p0[4]; A0.d[3] = p0[5];
  A1.d[0] = p1[0]; A1.d[1] = p1[1]; A1.d[2] = p1[4]; A1.d[3] = p1[5];
  union { v8i v; v4u q[2]; } B0, B1;
  const v4u* q0 = (const v4u*)(bRow0 + k0);
  const v4u* q1 = (const v4u*)(bRow1 + k0);
  B0.q[0] = q0[0]; B0.q[1] = q0[2];
  B1.q[0] = q1[0]; B1.q[1] = q1[2];
  f.a0 = A0.v; f.a1 = A1.v; f.b0 = B0.v; f.b1 = B1.v;
}

__device__ __forceinline__ void mma_slab(v8i acc[2][2], const Frag& f) {
  acc[0][0] = __builtin_amdgcn_wmma_i32_16x16x64_iu8(false, f.a0, false, f.b0, acc[0][0], false, false);
  acc[1][0] = __builtin_amdgcn_wmma_i32_16x16x64_iu8(false, f.a1, false, f.b0, acc[1][0], false, false);
  acc[0][1] = __builtin_amdgcn_wmma_i32_16x16x64_iu8(false, f.a0, false, f.b1, acc[0][1], false, false);
  acc[1][1] = __builtin_amdgcn_wmma_i32_16x16x64_iu8(false, f.a1, false, f.b1, acc[1][1], false, false);
}

__global__ __launch_bounds__(256) void gemm_wmma_iu8(
    const unsigned char* __restrict__ qw,
    const unsigned char* __restrict__ patches,
    const float* __restrict__ scales,
    const float* __restrict__ qwsum,
    const float* __restrict__ bias,
    float* __restrict__ out) {
  const int tid  = threadIdx.x;
  const int lane = tid & 31;
  const int wave = tid >> 5;
  const int half = lane >> 4;
  const int r    = lane & 15;
  const int obase = (wave >> 1) * 32;
  const int nbase = blockIdx.x * 64 + (wave & 1) * 32;

  const unsigned char* aRow0 = qw + (obase + r) * KDIM + half * 16;
  const unsigned char* aRow1 = qw + (obase + 16 + r) * KDIM + half * 16;
  const unsigned char* bRow0 = patches + (nbase + r) * KDIM + half * 16;
  const unsigned char* bRow1 = patches + (nbase + 16 + r) * KDIM + half * 16;

  v8i acc[2][2] = {};

  Frag f0, f1;
  load_slab(f0, aRow0, aRow1, bRow0, bRow1, 0);
#pragma unroll 1
  for (int k0 = 64; k0 < KDIM; k0 += 128) {
    load_slab(f1, aRow0, aRow1, bRow0, bRow1, k0);
    __builtin_prefetch(bRow0 + k0 + 128, 0, 1);
    __builtin_prefetch(bRow1 + k0 + 128, 0, 1);
    mma_slab(acc, f0);
    load_slab(f0, aRow0, aRow1, bRow0, bRow1, k0 + 64);
    mma_slab(acc, f1);
  }
  mma_slab(acc, f0);

  const float zf = scales[1];
  const float ss = scales[0] * scales[2];
#pragma unroll
  for (int i = 0; i < 2; ++i) {
#pragma unroll
    for (int j = 0; j < 8; ++j) {
      const int o = obase + i * 16 + half * 8 + j;
      const float corr = zf * qwsum[o];
      const float bo = bias[o];
#pragma unroll
      for (int t = 0; t < 2; ++t) {
        const int m = nbase + t * 16 + r;
        const int b = m / SPATIAL;
        const int s = m - b * SPATIAL;
        out[(b * O_OUT + o) * SPATIAL + s] = ((float)acc[i][t][j] - corr) * ss + bo;
      }
    }
  }
}

extern "C" void kernel_launch(void* const* d_in, const int* in_sizes, int n_in,
                              void* d_out, int out_size, void* d_ws, size_t ws_size,
                              hipStream_t stream) {
  const float* x    = (const float*)d_in[0];
  const float* w    = (const float*)d_in[1];
  const float* bias = (const float*)d_in[2];

  char* ws = (char*)d_ws;
  unsigned*      slots   = (unsigned*)ws;
  float*         scales  = (float*)(ws + 16);
  float*         qwsum   = (float*)(ws + 32);
  unsigned char* qw      = (unsigned char*)(ws + 576);
  unsigned char* patches = (unsigned char*)(ws + 576 + 73728);

  const int nx = BATCH * C_IN * H_IN * W_IN;
  const int nw = O_OUT * KDIM;
  const int npatch = MROWS * KDIM;

  init_slots<<<1, 32, 0, stream>>>(slots);
  minmax_kernel<<<512, 256, 0, stream>>>(x, nx, slots, 0);
  minmax_kernel<<<64, 256, 0, stream>>>(w, nw, slots, 2);
  compute_scales<<<1, 32, 0, stream>>>(slots, scales);
  quantize_weight<<<(nw + 255) / 256, 256, 0, stream>>>(w, scales, qw);
  qw_row_sum<<<O_OUT, 64, 0, stream>>>(qw, qwsum);
  im2col_quant<<<(npatch + 255) / 256, 256, 0, stream>>>(x, scales, patches);
  gemm_wmma_iu8<<<MROWS / 64, 256, 0, stream>>>(qw, patches, scales, qwsum, bias,
                                                (float*)d_out);
}
